// ScoreBlock_70059506532966
// MI455X (gfx1250) — matrix operand order repaired and checked by reading, not compiled, not run on hardware
//
#include <hip/hip_runtime.h>
#include <hip/hip_bf16.h>
#include <cstdint>
#include <cstddef>

// ---------------- problem constants ----------------
constexpr int cB  = 4;
constexpr int cNS = 1024;
constexpr int cNB = 4096;
constexpr int cC  = 256;
constexpr int cH  = 8;
constexpr int cD  = 32;          // head dim
constexpr int cNP = 64;
constexpr int cK  = cNS / 8;     // 128 top-k
constexpr float cSCALE = 0.17677669529663687f;   // 32^-0.5

// output layout (floats), concatenated in reference return order
constexpr size_t SEL_OFF = 0;                                  // [B,K,NS]
constexpr size_t IDX_OFF = SEL_OFF + (size_t)cB * cK * cNS;    // [B,K]
constexpr size_t PS_OFF  = IDX_OFF + (size_t)cB * cK;          // [B,NS]
constexpr size_t XS_OFF  = PS_OFF  + (size_t)cB * cNS;         // [B,NS,C]
constexpr size_t KER_OFF = XS_OFF  + (size_t)cB * cNS * cC;    // [B,C,1]

// ---------------- WMMA fragment types ----------------
typedef __attribute__((ext_vector_type(16))) _Float16 v16h;
typedef __attribute__((ext_vector_type(8)))  _Float16 v8h;
typedef __attribute__((ext_vector_type(8)))  float    v8f;

union Frag16 { v16h v; v8h h[2]; };

#define WMMA_F16(A, Bf, Cacc) \
  __builtin_amdgcn_wmma_f32_16x16x32_f16(false, (A), false, (Bf), (short)0, (Cacc), false, false)

// ---------------- DPP cross-lane reductions (VALU only, no LDS) ----------
// xor-butterfly over a 16-lane row with masks {1,2,7,15} (spans GF(2)^4):
//   quad_perm(1,0,3,2)=0xB1, quad_perm(2,3,0,1)=0x4E,
//   row_half_mirror=0x141 (xor7), row_mirror=0x140 (xor15)
template<int CTRL>
__device__ __forceinline__ float dpp_xf(float v) {
  return __int_as_float(
      __builtin_amdgcn_update_dpp(0, __float_as_int(v), CTRL, 0xF, 0xF, true));
}
__device__ __forceinline__ float rowmax16(float v) {
  v = fmaxf(v, dpp_xf<0xB1>(v));
  v = fmaxf(v, dpp_xf<0x4E>(v));
  v = fmaxf(v, dpp_xf<0x141>(v));
  v = fmaxf(v, dpp_xf<0x140>(v));
  return v;
}
__device__ __forceinline__ float rowsum16(float v) {
  v += dpp_xf<0xB1>(v);
  v += dpp_xf<0x4E>(v);
  v += dpp_xf<0x141>(v);
  v += dpp_xf<0x140>(v);
  return v;
}
__device__ __forceinline__ float allsum32(float v) {
  v = rowsum16(v);
  v += __shfl_xor(v, 16, 32);      // single cross-half combine
  return v;
}

// ======================================================================
// LayerNorm (fp32 in) -> f16 out. One wave per row of 256.
// ======================================================================
__global__ void __launch_bounds__(256)
ln_cvt(const float* __restrict__ x, const float* __restrict__ g,
       const float* __restrict__ b, _Float16* __restrict__ o, int nrows)
{
  int gw   = (blockIdx.x * blockDim.x + threadIdx.x) >> 5;
  int lane = threadIdx.x & 31;
  if (gw >= nrows) return;
  const float* row = x + (size_t)gw * cC;
  float vb0[8];
  float s = 0.f, ss = 0.f;
#pragma unroll
  for (int j = 0; j < 8; ++j) {
    float t = row[lane * 8 + j];
    vb0[j] = t; s += t; ss += t * t;
  }
  s  = allsum32(s);
  ss = allsum32(ss);
  float mean = s * (1.0f / cC);
  float var  = ss * (1.0f / cC) - mean * mean;
  float inv  = rsqrtf(var + 1e-5f);
#pragma unroll
  for (int j = 0; j < 8; ++j) {
    int c = lane * 8 + j;
    o[(size_t)gw * cC + c] = (_Float16)((vb0[j] - mean) * inv * g[c] + b[c]);
  }
}

// ======================================================================
// Weight fp32 [rows=Cin, ld=N] column slice -> f16 transposed [ncols, rows]
// ======================================================================
__global__ void __launch_bounds__(256)
cvt_w(const float* __restrict__ w, _Float16* __restrict__ o,
      int rows, int ldw, int col0, int ncols)
{
  int i = blockIdx.x * blockDim.x + threadIdx.x;
  if (i >= ncols * rows) return;
  int n = i / rows, c = i - n * rows;
  o[(size_t)n * rows + c] = (_Float16)w[(size_t)c * ldw + col0 + n];
}

// ======================================================================
// Generic WMMA GEMM: Y[M,N] = X_h[M,256] @ W (W stored transposed f16 [N,256])
// One wave computes a 16x16 f32 tile; K fully unrolled: 8x v_wmma.
// MODE 0: scatter to q/k  f16 [b][h][seq][32]     (seq = 1<<seqSh)
// MODE 1: scatter to v^T  f16 [b][h][32][seq]
// MODE 2/4: fp32 out = acc + bias[n] + resid
// MODE 3: f16 out = gelu_exact(acc + bias[n])
// ======================================================================
template<int MODE>
__global__ void __launch_bounds__(256)
gemm16(const _Float16* __restrict__ X, const _Float16* __restrict__ WT,
       const float* __restrict__ bias, const float* __restrict__ resid,
       void* __restrict__ outp, int M, int N, int seqSh)
{
  constexpr int Cin = cC;
  int gw   = (blockIdx.x * blockDim.x + threadIdx.x) >> 5;
  int lane = threadIdx.x & 31;
  int tilesN = N >> 4, tilesM = M >> 4;
  if (gw >= tilesM * tilesN) return;
  int tm = gw / tilesN, tn = gw - tm * tilesN;
  int hf = lane >> 4, l15 = lane & 15;
  const _Float16* xrow = X  + (size_t)(tm * 16 + l15) * Cin;
  const _Float16* wrow = WT + (size_t)(tn * 16 + l15) * Cin;
  v8f acc = {0.f, 0.f, 0.f, 0.f, 0.f, 0.f, 0.f, 0.f};
#pragma unroll
  for (int kk = 0; kk < Cin; kk += 32) {
    Frag16 a, bf;
    a.h[0]  = *(const v8h*)(xrow + kk + hf * 8);        // K = hf*8 .. +7
    a.h[1]  = *(const v8h*)(xrow + kk + hf * 8 + 16);   // K = 16+hf*8 .. +7
    bf.h[0] = *(const v8h*)(wrow + kk + hf * 8);        // K = hf*8 .. +7
    bf.h[1] = *(const v8h*)(wrow + kk + hf * 8 + 16);   // K = 16+hf*8 .. +7
    acc = WMMA_F16(a.v, bf.v, acc);
  }
  int n  = (tn << 4) + l15;
  int m0 = (tm << 4) + hf * 8;           // C/D row = hf*8 + r
  if (MODE == 0) {
    _Float16* o = (_Float16*)outp;
    int hh = n >> 5, dd = n & 31;
    int msk = (1 << seqSh) - 1;
#pragma unroll
    for (int r = 0; r < 8; ++r) {
      int m = m0 + r; int bb = m >> seqSh; int sq = m & msk;
      o[((((size_t)bb * cH + hh) << seqSh) + sq) * cD + dd] = (_Float16)acc[r];
    }
  } else if (MODE == 1) {
    _Float16* o = (_Float16*)outp;
    int hh = n >> 5, dd = n & 31;
    int msk = (1 << seqSh) - 1;
#pragma unroll
    for (int r = 0; r < 8; ++r) {
      int m = m0 + r; int bb = m >> seqSh; int sq = m & msk;
      o[((((size_t)bb * cH + hh) * cD + dd) << seqSh) + sq] = (_Float16)acc[r];
    }
  } else if (MODE == 2 || MODE == 4) {
    float* o = (float*)outp;
#pragma unroll
    for (int r = 0; r < 8; ++r) {
      int m = m0 + r;
      o[(size_t)m * cC + n] = acc[r] + bias[n] + resid[(size_t)m * cC + n];
    }
  } else { // MODE 3: exact GELU
    _Float16* o = (_Float16*)outp;
#pragma unroll
    for (int r = 0; r < 8; ++r) {
      int m = m0 + r;
      float t = acc[r] + bias[n];
      o[(size_t)m * cC + n] = (_Float16)(0.5f * t * (1.0f + erff(t * 0.70710678118f)));
    }
  }
}

// ======================================================================
// Flash attention. One wave per (b, h, 16-query tile). Streams keys in
// tiles of 32: 2 WMMAs for S = Q K^T (Kdim = D = 32), online softmax with
// DPP row-reductions (VALU only), P re-staged via per-wave LDS into an
// A-fragment, then 2 WMMAs for O += P V (against V^T, Kdim = 32 keys).
// ======================================================================
__global__ void __launch_bounds__(256)
attn_fa(const _Float16* __restrict__ q, const _Float16* __restrict__ k,
        const _Float16* __restrict__ vt, _Float16* __restrict__ o)
{
  __shared__ _Float16 Pst[8 * 16 * 32];
  int wib  = threadIdx.x >> 5;
  int gw   = (blockIdx.x * blockDim.x + threadIdx.x) >> 5;
  int lane = threadIdx.x & 31;
  int hf = lane >> 4, l15 = lane & 15;
  constexpr int QT = cNS / 16;
  int qt = gw % QT;
  int hh = (gw / QT) % cH;
  int bb = gw / (QT * cH);
  _Float16* pst = &Pst[wib * 16 * 32];

  const _Float16* qb = q  + (((size_t)bb * cH + hh) * cNS + (size_t)qt * 16) * cD;
  const _Float16* kb = k  + ((size_t)bb * cH + hh) * cNB * cD;
  const _Float16* vb = vt + ((size_t)bb * cH + hh) * cD * cNB;

  Frag16 aq;                                   // Q tile A-frag: row = l15, K = d
  aq.h[0] = *(const v8h*)(qb + l15 * cD + hf * 8);
  aq.h[1] = *(const v8h*)(qb + l15 * cD + hf * 8 + 16);

  float mi[8], li[8];
  v8f o0 = {0.f,0.f,0.f,0.f,0.f,0.f,0.f,0.f};
  v8f o1 = o0;
#pragma unroll
  for (int r = 0; r < 8; ++r) { mi[r] = -3.0e38f; li[r] = 0.f; }

  for (int kt = 0; kt < cNB; kt += 32) {
    // prefetch next key/value tiles into cache while this tile computes
    __builtin_prefetch(kb + (size_t)(kt + 32 + l15) * cD, 0, 3);
    __builtin_prefetch(vb + (size_t)l15 * cNB + kt + 32, 0, 3);

    v8f s0 = {0.f,0.f,0.f,0.f,0.f,0.f,0.f,0.f};
    v8f s1 = s0;
    Frag16 bk;
    const _Float16* kp = kb + (size_t)(kt + l15) * cD + hf * 8;   // col=key, K=d table order
    bk.h[0] = *(const v8h*)kp;  bk.h[1] = *(const v8h*)(kp + 16);
    s0 = WMMA_F16(aq.v, bk.v, s0);
    kp += 16 * cD;
    bk.h[0] = *(const v8h*)kp;  bk.h[1] = *(const v8h*)(kp + 16);
    s1 = WMMA_F16(aq.v, bk.v, s1);

#pragma unroll
    for (int r = 0; r < 8; ++r) {
      float a0 = s0[r] * cSCALE, a1 = s1[r] * cSCALE;
      float mx = rowmax16(fmaxf(a0, a1));
      float mnew  = fmaxf(mi[r], mx);
      float alpha = __expf(mi[r] - mnew);
      float p0 = __expf(a0 - mnew), p1 = __expf(a1 - mnew);
      float psum = rowsum16(p0 + p1);
      li[r] = li[r] * alpha + psum;
      o0[r] *= alpha;  o1[r] *= alpha;
      mi[r] = mnew;
      pst[(hf * 8 + r) * 32 + l15]      = (_Float16)p0;   // P row-major [qrow][key]
      pst[(hf * 8 + r) * 32 + 16 + l15] = (_Float16)p1;
    }
    asm volatile("s_wait_dscnt 0" ::: "memory");   // per-wave LDS RAW fence

    Frag16 ap, bv;                                 // P A-frag: row=l15, K=key-in-tile
    ap.h[0] = *(const v8h*)(pst + l15 * 32 + hf * 8);
    ap.h[1] = *(const v8h*)(pst + l15 * 32 + hf * 8 + 16);
    const _Float16* vp = vb + (size_t)l15 * cNB + kt + hf * 8;    // col=d, K=key table order
    bv.h[0] = *(const v8h*)vp;  bv.h[1] = *(const v8h*)(vp + 16);
    o0 = WMMA_F16(ap.v, bv.v, o0);                 // d = 0..15
    vp += (size_t)16 * cNB;
    bv.h[0] = *(const v8h*)vp;  bv.h[1] = *(const v8h*)(vp + 16);
    o1 = WMMA_F16(ap.v, bv.v, o1);                 // d = 16..31
  }

  _Float16* ob = o + ((size_t)bb * cNS + (size_t)qt * 16) * cC + hh * cD;
#pragma unroll
  for (int r = 0; r < 8; ++r) {
    float inv = 1.0f / li[r];
    int row = hf * 8 + r;
    ob[(size_t)row * cC + l15]      = (_Float16)(o0[r] * inv);
    ob[(size_t)row * cC + 16 + l15] = (_Float16)(o1[r] * inv);
  }
}

// ======================================================================
// kernels[b,c] = masked mean of x_b rows gathered by first NP/2 indices
// ======================================================================
__global__ void __launch_bounds__(256)
kernels_avg(const float* __restrict__ xb, const int* __restrict__ idxs,
            float* __restrict__ kout)
{
  int bb = blockIdx.x, c = threadIdx.x;
  int cnt = 0; float s = 0.f;
#pragma unroll 4
  for (int i = 0; i < cNP / 2; ++i) {
    int id = idxs[bb * cNP + i];
    if (id >= 0) { cnt++; s += xb[((size_t)bb * cNB + id) * cC + c]; }
  }
  float denom = (cnt > 0) ? (float)cnt : (float)(cNP / 2);
  kout[(size_t)bb * cC + c] = s / denom;
}

// ======================================================================
// pos_scores[b,n] = (cos(x_s[b,n], kernels[b]) + 1)/2 * has[b]; wave/row
// ======================================================================
__global__ void __launch_bounds__(256)
cos_scores(const float* __restrict__ xs, const float* __restrict__ ker,
           const int* __restrict__ idxs, float* __restrict__ ps)
{
  int gw   = (blockIdx.x * blockDim.x + threadIdx.x) >> 5;
  int lane = threadIdx.x & 31;
  if (gw >= cB * cNS) return;
  int bb = gw / cNS;
  const float* xr = xs + (size_t)gw * cC;
  const float* kr = ker + (size_t)bb * cC;
  float dot = 0.f, xn2 = 0.f, kn2 = 0.f;
#pragma unroll
  for (int j = 0; j < 8; ++j) {
    int c = lane * 8 + j;
    float xv = xr[c], kv = kr[c];
    dot += xv * kv; xn2 += xv * xv; kn2 += kv * kv;
  }
  dot = allsum32(dot);
  xn2 = allsum32(xn2);
  kn2 = allsum32(kn2);
  int cnt = 0;
#pragma unroll 4
  for (int i = 0; i < cNP / 2; ++i) cnt += (idxs[bb * cNP + i] >= 0) ? 1 : 0;
  float xn = fmaxf(sqrtf(xn2), 1e-8f);
  float kn = fmaxf(sqrtf(kn2), 1e-8f);
  float pos = (dot / (xn * kn) + 1.0f) * 0.5f * ((cnt > 0) ? 1.0f : 0.0f);
  if (lane == 0) ps[gw] = pos;
}

__global__ void __launch_bounds__(256)
zerof(float* __restrict__ p, int n)
{
  int i = blockIdx.x * blockDim.x + threadIdx.x;
  if (i < n) p[i] = 0.f;
}

// ======================================================================
// Exact top-K=128 of NS=1024 via stable rank count (matches lax.top_k
// ordering incl. index tie-break). One block per b; scores in LDS.
// selected is binary in forward: 1 where one-hot and score > 0.
// ======================================================================
__global__ void __launch_bounds__(256)
topk_sel(const float* __restrict__ ps, float* __restrict__ sel,
         float* __restrict__ idx_out)
{
  __shared__ float sc[cNS];
  int bb = blockIdx.x;
  for (int i = threadIdx.x; i < cNS; i += 256) sc[i] = ps[(size_t)bb * cNS + i];
  __syncthreads();
  for (int i = threadIdx.x; i < cNS; i += 256) {
    float v = sc[i];
    int rank = 0;
    for (int j = 0; j < cNS; ++j) {
      float u = sc[j];
      rank += (u > v) || (u == v && j < i);
    }
    if (rank < cK) {
      idx_out[(size_t)bb * cK + rank] = (float)i;
      if (v > 0.f) sel[((size_t)bb * cK + rank) * cNS + i] = 1.0f;
    }
  }
}

// ======================================================================
// host launcher
// ======================================================================
extern "C" void kernel_launch(void* const* d_in, const int* in_sizes, int n_in,
                              void* d_out, int out_size, void* d_ws, size_t ws_size,
                              hipStream_t stream)
{
  const float* x_b   = (const float*)d_in[0];
  const float* x_s   = (const float*)d_in[1];
  const int*   bidx  = (const int*)  d_in[2];
  const float* g1    = (const float*)d_in[3];
  const float* b1    = (const float*)d_in[4];
  const float* g2    = (const float*)d_in[5];
  const float* b2    = (const float*)d_in[6];
  const float* g3    = (const float*)d_in[7];
  const float* b3    = (const float*)d_in[8];
  const float* Wq    = (const float*)d_in[9];
  const float* Wkv   = (const float*)d_in[10];
  const float* Wproj = (const float*)d_in[11];
  const float* bproj = (const float*)d_in[12];
  const float* W1    = (const float*)d_in[13];
  const float* bb1   = (const float*)d_in[14];
  const float* W2    = (const float*)d_in[15];
  const float* bb2   = (const float*)d_in[16];
  float* out = (float*)d_out;

  char*  ws  = (char*)d_ws;
  size_t off = 0;
  auto alloc = [&](size_t bytes) -> char* {
    char* p = ws + off;
    off += (bytes + 255) & ~(size_t)255;
    return p;
  };
  _Float16* xs_n   = (_Float16*)alloc(2 * (size_t)cB * cNS * cC);
  _Float16* xb_n   = (_Float16*)alloc(2 * (size_t)cB * cNB * cC);
  _Float16* WqT    = (_Float16*)alloc(2 * (size_t)cC * cC);
  _Float16* WkT    = (_Float16*)alloc(2 * (size_t)cC * cC);
  _Float16* WvT    = (_Float16*)alloc(2 * (size_t)cC * cC);
  _Float16* WprojT = (_Float16*)alloc(2 * (size_t)cC * cC);
  _Float16* W1T    = (_Float16*)alloc(2 * (size_t)cC * cC);
  _Float16* W2T    = (_Float16*)alloc(2 * (size_t)cC * cC);
  _Float16* qbuf   = (_Float16*)alloc(2 * (size_t)cB * cH * cNS * cD);
  _Float16* kbuf   = (_Float16*)alloc(2 * (size_t)cB * cH * cNB * cD);
  _Float16* vTbuf  = (_Float16*)alloc(2 * (size_t)cB * cH * cD * cNB);
  _Float16* obuf   = (_Float16*)alloc(2 * (size_t)cB * cNS * cC);
  float*    xs2    = (float*)   alloc(4 * (size_t)cB * cNS * cC);
  _Float16* xs2ln  = (_Float16*)alloc(2 * (size_t)cB * cNS * cC);
  _Float16* hgelu  = (_Float16*)alloc(2 * (size_t)cB * cNS * cC);

  float* sel_o = out + SEL_OFF;
  float* idx_o = out + IDX_OFF;
  float* ps_o  = out + PS_OFF;
  float* xs_o  = out + XS_OFF;
  float* ker_o = out + KER_OFF;

  auto wblocks = [](int waves) { return (waves * 32 + 255) / 256; };

  // 1) LayerNorms -> f16
  ln_cvt<<<wblocks(cB * cNS), 256, 0, stream>>>(x_s, g1, b1, xs_n, cB * cNS);
  ln_cvt<<<wblocks(cB * cNB), 256, 0, stream>>>(x_b, g2, b2, xb_n, cB * cNB);

  // 2) weights -> f16 transposed [N, Cin]
  int wel = cC * cC, wblk = (wel + 255) / 256;
  cvt_w<<<wblk, 256, 0, stream>>>(Wq,    WqT,    cC, cC,     0, cC);
  cvt_w<<<wblk, 256, 0, stream>>>(Wkv,   WkT,    cC, 2 * cC, 0, cC);
  cvt_w<<<wblk, 256, 0, stream>>>(Wkv,   WvT,    cC, 2 * cC, cC, cC);
  cvt_w<<<wblk, 256, 0, stream>>>(Wproj, WprojT, cC, cC,     0, cC);
  cvt_w<<<wblk, 256, 0, stream>>>(W1,    W1T,    cC, cC,     0, cC);
  cvt_w<<<wblk, 256, 0, stream>>>(W2,    W2T,    cC, cC,     0, cC);

  // 3) projections (WMMA GEMMs); seqSh = log2(seq)
  int tq  = (cB * cNS / 16) * (cC / 16);
  int tkv = (cB * cNB / 16) * (cC / 16);
  gemm16<0><<<wblocks(tq),  256, 0, stream>>>(xs_n, WqT, nullptr, nullptr, qbuf,  cB * cNS, cC, 10);
  gemm16<0><<<wblocks(tkv), 256, 0, stream>>>(xb_n, WkT, nullptr, nullptr, kbuf,  cB * cNB, cC, 12);
  gemm16<1><<<wblocks(tkv), 256, 0, stream>>>(xb_n, WvT, nullptr, nullptr, vTbuf, cB * cNB, cC, 12);

  // 4) flash attention
  int awaves = cB * cH * (cNS / 16);
  attn_fa<<<wblocks(awaves), 256, 0, stream>>>(qbuf, kbuf, vTbuf, obuf);

  // 5) out-proj + residual (fp32), LN, MLP
  gemm16<2><<<wblocks(tq), 256, 0, stream>>>(obuf, WprojT, bproj, x_s, xs2, cB * cNS, cC, 0);
  ln_cvt<<<wblocks(cB * cNS), 256, 0, stream>>>(xs2, g3, b3, xs2ln, cB * cNS);
  gemm16<3><<<wblocks(tq), 256, 0, stream>>>(xs2ln, W1T, bb1, nullptr, hgelu, cB * cNS, cC, 0);
  gemm16<4><<<wblocks(tq), 256, 0, stream>>>(hgelu, W2T, bb2, xs2, xs_o, cB * cNS, cC, 0);

  // 6) scoring tail
  kernels_avg<<<cB, 256, 0, stream>>>(x_b, bidx, ker_o);
  cos_scores<<<wblocks(cB * cNS), 256, 0, stream>>>(xs_o, ker_o, bidx, ps_o);
  int nsel = cB * cK * cNS;
  zerof<<<(nsel + 255) / 256, 256, 0, stream>>>(sel_o, nsel);
  topk_sel<<<cB, 256, 0, stream>>>(ps_o, sel_o, idx_o);
}
